// DownSamplingGATBlock_66202625901215
// MI455X (gfx1250) — compile-verified, matrix operand order checked by reading, not run on hardware
//
#include <hip/hip_runtime.h>
#include <hip/hip_bf16.h>

// ---------------------------------------------------------------------------
// CDNA5 (gfx1250) GAT pipeline:
//   per layer: [WMMA bf16 GEMM] -> [node scores] -> [edge max] -> [edge exp/sum]
//              -> [edge scatter-add] -> [bias + relu]
// GEMM stages A row-major and B *transposed* (col-major tile) in LDS so both
// WMMA fragments load as two contiguous, 16B-aligned ds_load_b128 per lane.
// ---------------------------------------------------------------------------

typedef __attribute__((ext_vector_type(16))) __bf16 v16bf;
typedef __attribute__((ext_vector_type(8)))  float  v8f;

__device__ __forceinline__ unsigned int pack2bf(float a, float b) {
  // round-to-nearest-even f32 -> bf16, packed pair (a -> low half)
  unsigned int ua = __float_as_uint(a);
  ua = (ua + 0x7FFFu + ((ua >> 16) & 1u)) >> 16;
  unsigned int ub = __float_as_uint(b);
  ub = (ub + 0x7FFFu + ((ub >> 16) & 1u)) >> 16;
  return (ub << 16) | (ua & 0xFFFFu);
}

#define BM 128
#define BN 64
#define BK 32

// C = X[M,K] @ W[K,Ncols], fp32 in, bf16 WMMA, fp32 accumulate/out.
__global__ __launch_bounds__(256)
void gemm_wmma_bf16(const float* __restrict__ X, const float* __restrict__ W,
                    float* __restrict__ Hout, int M, int K, int Ncols) {
  // As: row-major [BM][BK] bf16 (packed as u32 pairs along K)
  // Bs: col-major [BN][BK] bf16 (packed as u32 pairs along K) -> transposed tile
  __shared__ unsigned int As32[BM * BK / 2];
  __shared__ unsigned int Bs32[BN * BK / 2];
  const __bf16* As = (const __bf16*)As32;
  const __bf16* Bs = (const __bf16*)Bs32;

  const int tid  = threadIdx.x;
  const int lane = tid & 31;
  const int wave = tid >> 5;                 // 0..7, each wave owns 16 rows x 64 cols
  const int rowTile = blockIdx.y * BM;
  const int colTile = blockIdx.x * BN;

  v8f acc[4] = {};

  const int mloc = lane & 15;
  const int khi  = (lane >> 4) << 3;         // lanes 16-31 cover K-offset +8 (ISA 7.12.2)

  for (int k0 = 0; k0 < K; k0 += BK) {
    // ---- stage A tile (128x32): float2 reads along K, packed bf16x2 stores
    for (int i = tid; i < BM * BK / 2; i += 256) {
      int r  = i >> 4;                       // BK/2 = 16 pairs per row
      int c  = (i & 15) << 1;
      int gr = rowTile + r;
      float f0 = 0.f, f1 = 0.f;
      if (gr < M) {
        const float2 v = *(const float2*)(X + (size_t)gr * K + (k0 + c));
        f0 = v.x; f1 = v.y;
        if (k0 + BK < K)
          __builtin_prefetch(X + (size_t)gr * K + (k0 + BK + c), 0, 0);
      }
      As32[i] = pack2bf(f0, f1);
    }
    // ---- stage B tile (32x64) transposed: coalesced col reads of two K-rows
    for (int i = tid; i < (BK / 2) * BN; i += 256) {
      int c  = i & (BN - 1);                 // column (consecutive lanes -> coalesced)
      int rp = i >> 6;                       // K-row pair index
      int r  = rp << 1;
      int gc = colTile + c;
      float f0 = 0.f, f1 = 0.f;
      if (gc < Ncols) {
        f0 = W[(size_t)(k0 + r)     * Ncols + gc];
        f1 = W[(size_t)(k0 + r + 1) * Ncols + gc];
      }
      Bs32[c * (BK / 2) + rp] = pack2bf(f0, f1);
    }
    __syncthreads();

    // A fragment: 16-bit A-matrix 16x32 layout
    // lane m = lane&15; element e<8 -> K=khi+e, e>=8 -> K=khi+e+8 (two 8-elem runs)
    v16bf afrag;
#pragma unroll
    for (int e = 0; e < 16; ++e) {
      int kk = khi + ((e < 8) ? e : (e + 8));
      afrag[e] = As[(wave * 16 + mloc) * BK + kk];
    }
#pragma unroll
    for (int g = 0; g < 4; ++g) {
      v16bf bfrag;                           // same two contiguous runs, col-major Bs
#pragma unroll
      for (int e = 0; e < 16; ++e) {
        int kk = khi + ((e < 8) ? e : (e + 8));
        bfrag[e] = Bs[(g * 16 + mloc) * BK + kk];
      }
      acc[g] = __builtin_amdgcn_wmma_f32_16x16x32_bf16(
          /*neg_a=*/false, afrag, /*neg_b=*/false, bfrag,
          /*c_mod=*/(short)0, acc[g], /*reuse_a=*/false, /*reuse_b=*/false);
    }
    __syncthreads();
  }

  // C/D layout: VGPR j -> M = j (+8 for lanes 16-31), N = lane&15
  const int rbase = rowTile + wave * 16 + khi;
#pragma unroll
  for (int g = 0; g < 4; ++g) {
    int col = colTile + g * 16 + mloc;
    if (col < Ncols) {
#pragma unroll
      for (int j = 0; j < 8; ++j) {
        int r = rbase + j;
        if (r < M) Hout[(size_t)r * Ncols + col] = acc[g][j];
      }
    }
  }
}

// s_src[n,h] = sum_c h[n,h,c]*a_src[h,c];  s_dst likewise
__global__ void node_scores(const float* __restrict__ Hbuf,
                            const float* __restrict__ a_src,
                            const float* __restrict__ a_dst,
                            float* __restrict__ Ssrc, float* __restrict__ Sdst,
                            int Nn, int H, int C) {
  int idx = blockIdx.x * blockDim.x + threadIdx.x;
  if (idx >= Nn * H) return;
  int n = idx / H, h = idx - n * H;
  const float* hp = Hbuf + (size_t)n * H * C + (size_t)h * C;
  const float* as = a_src + h * C;
  const float* ad = a_dst + h * C;
  float ss = 0.f, sd = 0.f;
  for (int c = 0; c < C; ++c) {
    float v = hp[c];
    ss += v * as[c];
    sd += v * ad[c];
  }
  Ssrc[idx] = ss;
  Sdst[idx] = sd;
}

__device__ __forceinline__ void edge_nodes(const int* __restrict__ ei, int E,
                                           int e, int& s, int& d) {
  if (e < E) { s = ei[e]; d = ei[E + e]; }
  else       { s = e - E; d = e - E; }      // appended self-loops
}

__device__ __forceinline__ unsigned int f32_ordered(float v) {
  unsigned int b = __float_as_uint(v);
  return (b & 0x80000000u) ? ~b : (b | 0x80000000u);
}
__device__ __forceinline__ float f32_unordered(unsigned int u) {
  unsigned int b = (u & 0x80000000u) ? (u & 0x7FFFFFFFu) : ~u;
  return __uint_as_float(b);
}

// pass 1: e = leaky_relu(ssrc[src]+sdst[dst]); segment-max via ordered-uint atomicMax
__global__ void edge_scores_max(const int* __restrict__ ei, int E, int Etot,
                                const float* __restrict__ Ssrc,
                                const float* __restrict__ Sdst,
                                float* __restrict__ Ebuf,
                                unsigned int* __restrict__ Mu, int H) {
  int e = blockIdx.x * blockDim.x + threadIdx.x;
  if (e >= Etot) return;
  int s, d; edge_nodes(ei, E, e, s, d);
  for (int h = 0; h < H; ++h) {
    float v = Ssrc[s * H + h] + Sdst[d * H + h];
    v = (v > 0.f) ? v : 0.2f * v;            // leaky_relu 0.2
    Ebuf[(size_t)e * H + h] = v;
    atomicMax(&Mu[d * H + h], f32_ordered(v));
  }
}

// pass 2: p = exp(e - m[dst]); denom[dst] += p
__global__ void edge_softmax_num(const int* __restrict__ ei, int E, int Etot,
                                 const unsigned int* __restrict__ Mu,
                                 float* __restrict__ Ebuf,
                                 float* __restrict__ Denom, int H) {
  int e = blockIdx.x * blockDim.x + threadIdx.x;
  if (e >= Etot) return;
  int s, d; edge_nodes(ei, E, e, s, d);
  (void)s;
  for (int h = 0; h < H; ++h) {
    float m = f32_unordered(Mu[d * H + h]);
    float p = __expf(Ebuf[(size_t)e * H + h] - m);
    Ebuf[(size_t)e * H + h] = p;
    atomicAdd(&Denom[d * H + h], p);
  }
}

// pass 3: out[dst] += (p/denom[dst]) * h[src]  (float4 chunks; L2-resident working set)
__global__ void edge_aggregate(const int* __restrict__ ei, int E, int Etot,
                               const float* __restrict__ Ebuf,
                               const float* __restrict__ Denom,
                               const float* __restrict__ Hbuf,
                               float* __restrict__ Out, int H, int C) {
  const int F = H * C;
  const int nchunk = F >> 2;                 // float4 chunks (C % 4 == 0)
  int idx = blockIdx.x * blockDim.x + threadIdx.x;
  if (idx >= Etot * nchunk) return;
  int e = idx / nchunk;
  int ch = idx - e * nchunk;
  int f = ch << 2;
  int h = f / C;
  int s, d; edge_nodes(ei, E, e, s, d);
  float alpha = Ebuf[(size_t)e * H + h] / Denom[d * H + h];
  const float4 hv = *(const float4*)(Hbuf + (size_t)s * F + f);
  float* op = Out + (size_t)d * F + f;
  atomicAdd(op + 0, alpha * hv.x);
  atomicAdd(op + 1, alpha * hv.y);
  atomicAdd(op + 2, alpha * hv.z);
  atomicAdd(op + 3, alpha * hv.w);
}

__global__ void bias_act(const float* __restrict__ In, const float* __restrict__ b,
                         float* __restrict__ Outp, int Nn, int F, int do_relu) {
  int idx = blockIdx.x * blockDim.x + threadIdx.x;
  if (idx >= Nn * F) return;
  int f = idx % F;
  float v = In[idx] + b[f];
  if (do_relu) v = fmaxf(v, 0.f);
  Outp[idx] = v;
}

// ---------------------------------------------------------------------------

static inline int cdiv(long long a, long long b) { return (int)((a + b - 1) / b); }

extern "C" void kernel_launch(void* const* d_in, const int* in_sizes, int n_in,
                              void* d_out, int out_size, void* d_ws, size_t ws_size,
                              hipStream_t stream) {
  const float* x      = (const float*)d_in[0];
  const int*   ei     = (const int*)  d_in[1];   // [2, E]
  const float* W1     = (const float*)d_in[2];
  const float* a_src1 = (const float*)d_in[3];
  const float* a_dst1 = (const float*)d_in[4];
  const float* b1     = (const float*)d_in[5];
  const float* W2     = (const float*)d_in[6];
  const float* a_src2 = (const float*)d_in[7];
  const float* a_dst2 = (const float*)d_in[8];
  const float* b2     = (const float*)d_in[9];
  const float* W3     = (const float*)d_in[10];
  const float* a_src3 = (const float*)d_in[11];
  const float* a_dst3 = (const float*)d_in[12];
  const float* b3     = (const float*)d_in[13];

  const int IN_F = 1024;
  const int Nn   = in_sizes[0] / IN_F;         // 50000
  const int Ee   = in_sizes[1] / 2;            // 800000
  const int Etot = Ee + Nn;                    // + self loops
  const int Hmax = 8;

  // ---- workspace carve-out (256B aligned) ----
  char* ws = (char*)d_ws;
  size_t off = 0;
  auto carve = [&](size_t bytes) -> void* {
    void* p = ws + off;
    off += (bytes + 255) & ~(size_t)255;
    return p;
  };
  float*        B0    = (float*)carve((size_t)Nn * 512 * sizeof(float));   // h buffer
  float*        B1    = (float*)carve((size_t)Nn * 512 * sizeof(float));   // out / act buffer
  float*        Ssrc  = (float*)carve((size_t)Nn * Hmax * sizeof(float));
  float*        Sdst  = (float*)carve((size_t)Nn * Hmax * sizeof(float));
  unsigned int* Mu    = (unsigned int*)carve((size_t)Nn * Hmax * sizeof(unsigned int));
  float*        Denom = (float*)carve((size_t)Nn * Hmax * sizeof(float));
  float*        Ebuf  = (float*)carve((size_t)Etot * Hmax * sizeof(float));
  (void)ws_size;

  const int TB = 256;

  auto run_layer = [&](const float* Xin, int Fin,
                       const float* Wm, const float* as, const float* ad,
                       const float* bias, int H, int C,
                       float* Hbuf, float* Obuf, float* Act, int do_relu) {
    const int F = H * C;
    // GEMM: Hbuf = Xin @ Wm
    dim3 gg(cdiv(F, BN), cdiv(Nn, BM));
    gemm_wmma_bf16<<<gg, TB, 0, stream>>>(Xin, Wm, Hbuf, Nn, Fin, F);
    // attention scores
    node_scores<<<cdiv((long long)Nn * H, TB), TB, 0, stream>>>(
        Hbuf, as, ad, Ssrc, Sdst, Nn, H, C);
    // reset segment state + output accumulator
    hipMemsetAsync(Mu,    0, (size_t)Nn * H * sizeof(unsigned int), stream);
    hipMemsetAsync(Denom, 0, (size_t)Nn * H * sizeof(float),        stream);
    hipMemsetAsync(Obuf,  0, (size_t)Nn * F * sizeof(float),        stream);
    // edge passes
    edge_scores_max<<<cdiv(Etot, TB), TB, 0, stream>>>(ei, Ee, Etot, Ssrc, Sdst, Ebuf, Mu, H);
    edge_softmax_num<<<cdiv(Etot, TB), TB, 0, stream>>>(ei, Ee, Etot, Mu, Ebuf, Denom, H);
    edge_aggregate<<<cdiv((long long)Etot * (F >> 2), TB), TB, 0, stream>>>(
        ei, Ee, Etot, Ebuf, Denom, Hbuf, Obuf, H, C);
    // bias (+relu) -> activations for next layer / final output
    bias_act<<<cdiv((long long)Nn * F, TB), TB, 0, stream>>>(Obuf, bias, Act, Nn, F, do_relu);
  };

  // Layer 1: 1024 -> 8x64 = 512, concat, relu.  h1=B0, out1=B1, act1=B1 (in place)
  run_layer(x, IN_F, W1, a_src1, a_dst1, b1, 8, 64, B0, B1, B1, 1);

  // Layer 2: 512 -> 8x32 = 256.  x2=B1, h2=B0[0:], out2=B0[+Nn*256], act2=B1 (reuse)
  float* out2 = B0 + (size_t)Nn * 256;
  run_layer(B1, 512, W2, a_src2, a_dst2, b2, 8, 32, B0, out2, B1, 1);

  // Layer 3: 256 -> 1x32 = 32.  x3=B1, h3=B0[0:], out3=B0[+Nn*256], final -> d_out (no relu)
  float* out3 = B0 + (size_t)Nn * 256;
  run_layer(B1, 256, W3, a_src3, a_dst3, b3, 1, 32, B0, out3, (float*)d_out, 0);
}
